// SimpleRetention_38173669327185
// MI455X (gfx1250) — matrix operand order repaired and checked by reading, not compiled, not run on hardware
//
#include <hip/hip_runtime.h>
#include <hip/hip_bf16.h>

// Problem constants (from reference)
#define B_  8
#define S_  2048
#define E_  1024
#define H_  128

typedef __attribute__((ext_vector_type(16))) __bf16 v16bf;
typedef __attribute__((ext_vector_type(8)))  __bf16 v8bf;
typedef __attribute__((ext_vector_type(8)))  float  v8f;

static __device__ __forceinline__ v8f wmma_bf16(v16bf a, v16bf b, v8f c) {
  // D = A(16x32 bf16) * B(32x16 bf16) + C(16x16 f32)
  return __builtin_amdgcn_wmma_f32_16x16x32_bf16(
      /*neg_a=*/false, a, /*neg_b=*/false, b,
      /*c_mod=*/(short)0, c, /*reuse_a=*/false, /*reuse_b=*/false);
}

// A operand (16x32, MxK): lane holds row M=lane%16; K = 16*lh + 0..15
// contiguous (lh = lane/16), matching the B operand order.
static __device__ __forceinline__ v16bf load_a_bf16(const __bf16* base, int lh) {
  v8bf c0 = *(const v8bf*)(base + 16 * lh);
  v8bf c1 = *(const v8bf*)(base + 16 * lh + 8);
  v16bf a;
#pragma unroll
  for (int t = 0; t < 8; ++t) { a[t] = c0[t]; a[t + 8] = c1[t]; }
  return a;
}

static __device__ __forceinline__ v16bf load_a_f32(const float* base, int lh) {
  const float* p0 = base + 16 * lh;
  const float* p1 = base + 16 * lh + 8;
  v16bf a;
#pragma unroll
  for (int t = 0; t < 8; ++t) { a[t] = (__bf16)p0[t]; a[t + 8] = (__bf16)p1[t]; }
  return a;
}

// B operand (32x16, KxN): lane holds column N=lane%16, K = 16*lh + 0..15 contiguous.
static __device__ __forceinline__ v16bf load_b_bf16(const __bf16* colbase, int lh) {
  return *(const v16bf*)(colbase + 16 * lh);
}

// ---------------------------------------------------------------------------
// Kernel 0: W (E x H, f32) -> W^T (H x E, bf16) for contiguous B-operand loads
// ---------------------------------------------------------------------------
__global__ __launch_bounds__(256) void k_wcvt(
    const float* __restrict__ WQ, const float* __restrict__ WK,
    const float* __restrict__ WV,
    __bf16* __restrict__ WtQ, __bf16* __restrict__ WtK, __bf16* __restrict__ WtV) {
  const float* src = (blockIdx.y == 0) ? WQ : (blockIdx.y == 1) ? WK : WV;
  __bf16* dst = (blockIdx.y == 0) ? WtQ : (blockIdx.y == 1) ? WtK : WtV;
  int idx = blockIdx.x * 256 + threadIdx.x;  // 0 .. E_*H_-1
  int k = idx >> 7;                          // embed index
  int n = idx & (H_ - 1);                    // head index
  dst[n * E_ + k] = (__bf16)src[idx];
}

// ---------------------------------------------------------------------------
// Kernel 1: projections.  X (B*S x E, f32) @ W (E x H) -> bf16.
// sel 0: query -> Qbf (B*S x H), sel 1: key -> Kbf (B*S x H),
// sel 2: value -> Vt stored transposed (B x H x S) for the 2nd attention GEMM.
// Block = 256 thr (8 waves); block computes a 128(M) x 128(N) tile, K=1024.
// Software-pipelined: A + 8 B-tiles double-buffered in registers so the next
// k-step's loads overlap the current 8 WMMAs.
// ---------------------------------------------------------------------------
__global__ __launch_bounds__(256) void k_proj(
    const float* __restrict__ query, const float* __restrict__ key,
    const float* __restrict__ value,
    const __bf16* __restrict__ WtQ, const __bf16* __restrict__ WtK,
    const __bf16* __restrict__ WtV,
    __bf16* __restrict__ Qbf, __bf16* __restrict__ Kbf, __bf16* __restrict__ Vt) {
  const int sel = blockIdx.y;
  const float*  X  = (sel == 0) ? query : (sel == 1) ? key : value;
  const __bf16* Wt = (sel == 0) ? WtQ : (sel == 1) ? WtK : WtV;
  __bf16* Out = (sel == 0) ? Qbf : (sel == 1) ? Kbf : Vt;

  const int w  = threadIdx.x >> 5;  // wave in block (0..7): owns 16 M rows
  const int l  = threadIdx.x & 31;
  const int lm = l & 15;
  const int lh = l >> 4;
  const long mb = blockIdx.x;

  const float* Arow = X + (mb * 128 + w * 16 + lm) * (long)E_;

  v8f zero = {};
  v8f acc[8];
#pragma unroll
  for (int t = 0; t < 8; ++t) acc[t] = zero;

  // prologue: k-step 0 operands
  v16bf a_cur = load_a_f32(Arow, lh);
  v16bf b_cur[8];
#pragma unroll
  for (int nt = 0; nt < 8; ++nt)
    b_cur[nt] = load_b_bf16(Wt + (long)(nt * 16 + lm) * E_, lh);

#pragma unroll 2
  for (int ks = 0; ks < E_ / 32; ++ks) {
    // issue next k-step's loads (redundant reload of step 0 on the last
    // iteration; harmless, keeps the loop branch-free)
    const int kn = (ks + 1 < E_ / 32) ? (ks + 1) * 32 : 0;
    v16bf a_nxt = load_a_f32(Arow + kn, lh);
    v16bf b_nxt[8];
#pragma unroll
    for (int nt = 0; nt < 8; ++nt)
      b_nxt[nt] = load_b_bf16(Wt + (long)(nt * 16 + lm) * E_ + kn, lh);

    // compute on current buffers while the loads fly
#pragma unroll
    for (int nt = 0; nt < 8; ++nt)
      acc[nt] = wmma_bf16(a_cur, b_cur[nt], acc[nt]);

    a_cur = a_nxt;
#pragma unroll
    for (int nt = 0; nt < 8; ++nt) b_cur[nt] = b_nxt[nt];
  }

  // C/D layout: VGPR r, lane -> M = r + 8*lh, N = lm   (ISA 7.12.2)
#pragma unroll
  for (int nt = 0; nt < 8; ++nt) {
#pragma unroll
    for (int r = 0; r < 8; ++r) {
      const long m = mb * 128 + w * 16 + r + 8 * lh;  // flat over B*S
      const int  n = nt * 16 + lm;
      const __bf16 val = (__bf16)acc[nt][r];
      if (sel == 2) {
        const long bb = m >> 11;        // / S_
        const long s  = m & (S_ - 1);
        Out[(bb * H_ + n) * (long)S_ + s] = val;  // V^T: (B, H, S)
      } else {
        Out[m * (long)H_ + n] = val;              // (B*S, H)
      }
    }
  }
}

// ---------------------------------------------------------------------------
// Kernel 2: fused causal retention: O = tril(Q K^T) V, never materializing SxS.
// Grid (S/128, B); block = 256 thr = 8 waves; wave w owns query rows
// [i*128 + 16w, +16).  Per key-block j<=i: P strip (16x128) via WMMA,
// diagonal mask, LDS round-trip to A-layout, then O += P @ V via WMMA.
// B-tiles of each k-step are gathered into register arrays so the 16 b128
// loads issue as a clause and overlap the previous step's WMMAs; the next
// key block's K/V tiles are prefetched into WGP$.
// ---------------------------------------------------------------------------
__global__ __launch_bounds__(256) void k_attn(
    const __bf16* __restrict__ Qbf, const __bf16* __restrict__ Kbf,
    const __bf16* __restrict__ Vt, float* __restrict__ out) {
  __shared__ __bf16 lds[8][16][144];  // per-wave 16x128 P strip, padded rows

  const int w  = threadIdx.x >> 5;
  const int l  = threadIdx.x & 31;
  const int lm = l & 15;
  const int lh = l >> 4;
  const int i  = blockIdx.x;  // query block of 128 rows
  const int b  = blockIdx.y;

  const __bf16* Qrow = Qbf + ((long)b * S_ + i * 128 + w * 16 + lm) * H_;

  v8f zero = {};
  v8f o[8];
#pragma unroll
  for (int t = 0; t < 8; ++t) o[t] = zero;

  for (int j = 0; j <= i; ++j) {
    // ---- prefetch next key block's K and V tiles into WGP$ ----
    if (j < i) {
      // K block j+1: 128 rows x 256 B; lanes cover it at 1 KB granularity
      __builtin_prefetch(
          Kbf + ((long)b * S_ + (j + 1) * 128 + l * 4) * H_, 0, 3);
      // V^T block j+1: 128 rows, 256 B span each
      __builtin_prefetch(
          Vt + ((long)b * H_ + l * 4) * S_ + (j + 1) * 128, 0, 3);
    }

    // ---- P = Q_strip (16xH) @ K_block^T (Hx128) ----
    v8f p[8];
#pragma unroll
    for (int t = 0; t < 8; ++t) p[t] = zero;

#pragma unroll
    for (int ks = 0; ks < 4; ++ks) {  // H=128 in steps of 32
      const int h0 = ks * 32;
      v16bf a = load_a_bf16(Qrow + h0, lh);
      v16bf bt[8];
#pragma unroll
      for (int nt = 0; nt < 8; ++nt)  // batch the 16 b128 loads
        bt[nt] = load_b_bf16(
            Kbf + ((long)b * S_ + j * 128 + nt * 16 + lm) * H_ + h0, lh);
#pragma unroll
      for (int nt = 0; nt < 8; ++nt)
        p[nt] = wmma_bf16(a, bt[nt], p[nt]);
    }

    // ---- causal mask on the diagonal block ----
    if (j == i) {
#pragma unroll
      for (int nt = 0; nt < 8; ++nt) {
#pragma unroll
        for (int r = 0; r < 8; ++r) {
          const int m_loc = w * 16 + r + 8 * lh;  // query pos within 128-block
          const int k_loc = nt * 16 + lm;         // key pos within 128-block
          if (k_loc > m_loc) p[nt][r] = 0.0f;
        }
      }
    }

    // ---- re-layout P: C/D layout -> A layout via per-wave LDS ----
#pragma unroll
    for (int nt = 0; nt < 8; ++nt) {
#pragma unroll
      for (int r = 0; r < 8; ++r)
        lds[w][r + 8 * lh][nt * 16 + lm] = (__bf16)p[nt][r];
    }
    // same-wave cross-lane RAW through LDS: LDS is in-order per wave, but wait
    // explicitly and fence the compiler.
    asm volatile("s_wait_dscnt 0x0" ::: "memory");

    // ---- O += P (16x128 over keys) @ V_block (128 x H) ----
    const __bf16* lrow = &lds[w][lm][0];
#pragma unroll
    for (int ks = 0; ks < 4; ++ks) {  // 128 keys in steps of 32
      const int c0 = ks * 32;
      v16bf a2 = load_a_bf16(lrow + c0, lh);
      v16bf bt[8];
#pragma unroll
      for (int nt = 0; nt < 8; ++nt)
        bt[nt] = load_b_bf16(
            Vt + ((long)b * H_ + nt * 16 + lm) * S_ + j * 128 + c0, lh);
#pragma unroll
      for (int nt = 0; nt < 8; ++nt)
        o[nt] = wmma_bf16(a2, bt[nt], o[nt]);
    }
    asm volatile("" ::: "memory");  // next iter's LDS stores stay after loads
  }

  // ---- store O (f32, row-major (B,S,H)) ----
#pragma unroll
  for (int nt = 0; nt < 8; ++nt) {
#pragma unroll
    for (int r = 0; r < 8; ++r) {
      const long m = (long)b * S_ + i * 128 + w * 16 + r + 8 * lh;
      out[m * (long)H_ + nt * 16 + lm] = o[nt][r];
    }
  }
}

// ---------------------------------------------------------------------------
extern "C" void kernel_launch(void* const* d_in, const int* in_sizes, int n_in,
                              void* d_out, int out_size, void* d_ws, size_t ws_size,
                              hipStream_t stream) {
  (void)in_sizes; (void)n_in; (void)out_size; (void)ws_size;
  // setup_inputs() dict order: key, query, value, W_Q, W_K, W_V
  const float* key   = (const float*)d_in[0];
  const float* query = (const float*)d_in[1];
  const float* value = (const float*)d_in[2];
  const float* W_Q   = (const float*)d_in[3];
  const float* W_K   = (const float*)d_in[4];
  const float* W_V   = (const float*)d_in[5];

  // Workspace layout (bf16): 3x W^T (H*E) + Qbf + Kbf + Vt  (~12.75 MB)
  __bf16* WtQ = (__bf16*)d_ws;
  __bf16* WtK = WtQ + (long)H_ * E_;
  __bf16* WtV = WtK + (long)H_ * E_;
  __bf16* Qbf = WtV + (long)H_ * E_;
  __bf16* Kbf = Qbf + (long)B_ * S_ * H_;
  __bf16* Vt  = Kbf + (long)B_ * S_ * H_;

  k_wcvt<<<dim3((E_ * H_) / 256, 3), 256, 0, stream>>>(W_Q, W_K, W_V,
                                                       WtQ, WtK, WtV);
  k_proj<<<dim3((B_ * S_) / 128, 3), 256, 0, stream>>>(query, key, value,
                                                       WtQ, WtK, WtV,
                                                       Qbf, Kbf, Vt);
  k_attn<<<dim3(S_ / 128, B_), 256, 0, stream>>>(Qbf, Kbf, Vt, (float*)d_out);
}
